// LSTMScratchCell_1090921693307
// MI455X (gfx1250) — matrix operand order repaired and checked by reading, not compiled, not run on hardware
//
#include <hip/hip_runtime.h>
#include <hip/hip_bf16.h>

#define BATCH 65536
#define HID   256
#define KDIM  512          // 256 (x) + 256 (h_prev)
#define LDS_STRIDE 536     // 512 + 24 pad: 16B-aligned rows, conflict-free b128 reads

typedef __attribute__((ext_vector_type(16))) __bf16          bf16x16;
typedef __attribute__((ext_vector_type(8)))  float           floatx8;
typedef __attribute__((ext_vector_type(4)))  float           f32x4;   // NT-loadable
typedef __attribute__((ext_vector_type(4)))  unsigned short  u16x4;

struct Pack32 { uint4 a, b; };

__device__ __forceinline__ bf16x16 frag_cast(uint4 a, uint4 b) {
  Pack32 p{a, b};
  return __builtin_bit_cast(bf16x16, p);
}

__device__ __forceinline__ unsigned short f32_to_bf16_rne(float f) {
  unsigned int u = __builtin_bit_cast(unsigned int, f);
  u += 0x7FFFu + ((u >> 16) & 1u);
  return (unsigned short)(u >> 16);
}

__device__ __forceinline__ float fast_sigmoid(float x) {
  return __builtin_amdgcn_rcpf(1.0f + __expf(-x));
}
__device__ __forceinline__ float fast_tanh(float x) {
  // tanh(x) = 2*sigmoid(2x) - 1
  return 2.0f * fast_sigmoid(2.0f * x) - 1.0f;
}

// ---------------------------------------------------------------------------
// One-time (per launch) weight repack: fp32 [K=256][H=256] x 8 matrices
// -> bf16 Wt[col = gate*256 + j][k = 0..511]  (K-major: B fragments contiguous)
// ---------------------------------------------------------------------------
__global__ __launch_bounds__(256)
void convert_weights(const float* __restrict__ Wxi, const float* __restrict__ Whi,
                     const float* __restrict__ Wxf, const float* __restrict__ Whf,
                     const float* __restrict__ Wxo, const float* __restrict__ Who,
                     const float* __restrict__ Wxc, const float* __restrict__ Whc,
                     unsigned short* __restrict__ Wt) {
  const int idx = blockIdx.x * 256 + threadIdx.x;   // over 1024*512
  const int col = idx >> 9;                         // 0..1023 (uniform per block)
  const int k   = idx & 511;                        // 0..511
  const int gate = col >> 8;                        // 0..3 (i,f,o,c) - uniform
  const int j    = col & 255;
  const float* wx = (gate == 0) ? Wxi : (gate == 1) ? Wxf : (gate == 2) ? Wxo : Wxc;
  const float* wh = (gate == 0) ? Whi : (gate == 1) ? Whf : (gate == 2) ? Who : Whc;
  const float v = (k < 256) ? wx[k * 256 + j] : wh[(k - 256) * 256 + j];
  Wt[idx] = f32_to_bf16_rne(v);
}

// ---------------------------------------------------------------------------
// Fused LSTM cell. Workgroup = 32 batch rows x ALL 256 hidden cols:
// the A stripe (32x512 bf16 in LDS) is loaded from HBM exactly once and
// reused across a 4-iteration hidden-column loop (16 col-tiles / 4 col-waves).
// Each wave owns a 16x16 hidden tile per iteration with 4 gate accumulators,
// so the LSTM nonlinear epilogue is wave-local (gates never hit memory).
// ---------------------------------------------------------------------------
__global__ __launch_bounds__(256)
void lstm_wmma_kernel(const float* __restrict__ x,
                      const float* __restrict__ h_prev,
                      const float* __restrict__ c_prev,
                      const unsigned short* __restrict__ Wt,  // bf16 [1024][512]
                      const float* __restrict__ b_i, const float* __restrict__ b_f,
                      const float* __restrict__ b_o, const float* __restrict__ b_c,
                      float* __restrict__ out_h, float* __restrict__ out_c) {
  __shared__ unsigned short As[32][LDS_STRIDE];   // 32 rows x 512 K, bf16

  const int m0  = blockIdx.x * 32;   // row stripe
  const int tid = threadIdx.x;

  // ---- Stage A = [x | h_prev] (32 rows x 512 K) as bf16 into LDS ----------
  // Streaming data: nontemporal so it doesn't evict the L2-resident weights.
  {
    const int rr = tid >> 3;           // 0..31
    const int kc = (tid & 7) << 6;     // 0,64,...,448 (never straddles x/h split)
    const size_t rowoff = (size_t)(m0 + rr) * HID;
    const f32x4* src = (const f32x4*)((kc < HID)
                           ? (x + rowoff + kc)
                           : (h_prev + rowoff + (kc - HID)));
#pragma unroll
    for (int i = 0; i < 16; ++i) {
      f32x4 f = __builtin_nontemporal_load(src + i);
      u16x4 p;
      p.x = f32_to_bf16_rne(f.x);
      p.y = f32_to_bf16_rne(f.y);
      p.z = f32_to_bf16_rne(f.z);
      p.w = f32_to_bf16_rne(f.w);
      *(u16x4*)&As[rr][kc + i * 4] = p;
    }
  }
  __syncthreads();

  const int wave    = tid >> 5;
  const int lane    = tid & 31;
  const int rblk    = wave & 1;        // 16-row block within stripe
  const int cquad   = wave >> 1;       // which 64-col quadrant this wave walks
  const int halfsel = lane >> 4;       // hi/lo half of wave
  const int l15     = lane & 15;

  // A fragment source: lane<16 -> row l15, K {ks..ks+7, ks+16..23};
  //                    lane>=16 -> row l15, K {ks+8..15, ks+24..31}
  const unsigned short* arow = &As[rblk * 16 + l15][0];
  const int rbase = m0 + rblk * 16 + halfsel * 8;

#pragma unroll 1
  for (int iter = 0; iter < 4; ++iter) {
    const int hcol0 = (cquad * 4 + iter) * 16;   // 16 hidden cols this pass
    // B fragment source: lane n holds column (gate*256 + hcol0 + l15), K-major.
    const unsigned short* bbase =
        Wt + (size_t)(hcol0 + l15) * KDIM + halfsel * 8;

    floatx8 acc0 = {0.f, 0.f, 0.f, 0.f, 0.f, 0.f, 0.f, 0.f};
    floatx8 acc1 = acc0, acc2 = acc0, acc3 = acc0;

#pragma unroll
    for (int ks = 0; ks < KDIM; ks += 32) {
      const uint4* ap = (const uint4*)(arow + ks + halfsel * 8);
      bf16x16 afrag = frag_cast(ap[0], ap[2]);   // ds_load_b128 x2

      const unsigned short* bp = bbase + ks;
      const uint4* bq0 = (const uint4*)(bp);                  // gate i
      const uint4* bq1 = (const uint4*)(bp + 1 * 256 * KDIM); // gate f
      const uint4* bq2 = (const uint4*)(bp + 2 * 256 * KDIM); // gate o
      const uint4* bq3 = (const uint4*)(bp + 3 * 256 * KDIM); // gate c
      bf16x16 bf0 = frag_cast(bq0[0], bq0[2]);   // global_load_b128 x2 each
      bf16x16 bf1 = frag_cast(bq1[0], bq1[2]);
      bf16x16 bf2 = frag_cast(bq2[0], bq2[2]);
      bf16x16 bf3 = frag_cast(bq3[0], bq3[2]);

      acc0 = __builtin_amdgcn_wmma_f32_16x16x32_bf16(false, afrag, false, bf0,
                                                     (short)0, acc0, false, false);
      acc1 = __builtin_amdgcn_wmma_f32_16x16x32_bf16(false, afrag, false, bf1,
                                                     (short)0, acc1, false, false);
      acc2 = __builtin_amdgcn_wmma_f32_16x16x32_bf16(false, afrag, false, bf2,
                                                     (short)0, acc2, false, false);
      acc3 = __builtin_amdgcn_wmma_f32_16x16x32_bf16(false, afrag, false, bf3,
                                                     (short)0, acc3, false, false);
    }

    // ---- Epilogue: wave-local LSTM nonlinearities + state update ----------
    // C/D layout: VGPR v -> M = v + 8*halfsel, N = l15.
    const int   col = hcol0 + l15;
    const float bi  = b_i[col];
    const float bfv = b_f[col];
    const float bo  = b_o[col];
    const float bc  = b_c[col];

#pragma unroll
    for (int v = 0; v < 8; ++v) {
      const int row = rbase + v;
      const size_t off = (size_t)row * HID + col;
      const float gi = acc0[v] + bi;
      const float gf = acc1[v] + bfv;
      const float go = acc2[v] + bo;
      const float gc = acc3[v] + bc;
      const float ig = fast_sigmoid(gi);
      const float fg = fast_sigmoid(gf);
      const float og = fast_sigmoid(go);
      const float ct = fast_tanh(gc);
      const float c  = fg * __builtin_nontemporal_load(c_prev + off) + ig * ct;
      const float h  = og * fast_tanh(c);
      __builtin_nontemporal_store(h, out_h + off);
      __builtin_nontemporal_store(c, out_c + off);
    }
  }
}

// ---------------------------------------------------------------------------
extern "C" void kernel_launch(void* const* d_in, const int* in_sizes, int n_in,
                              void* d_out, int out_size, void* d_ws, size_t ws_size,
                              hipStream_t stream) {
  // setup_inputs() dict order:
  // x, h_prev, c_prev, W_xi, W_hi, W_xf, W_hf, W_xo, W_ho, W_xc, W_hc,
  // b_i, b_f, b_o, b_c
  const float* x      = (const float*)d_in[0];
  const float* h_prev = (const float*)d_in[1];
  const float* c_prev = (const float*)d_in[2];
  const float* W_xi   = (const float*)d_in[3];
  const float* W_hi   = (const float*)d_in[4];
  const float* W_xf   = (const float*)d_in[5];
  const float* W_hf   = (const float*)d_in[6];
  const float* W_xo   = (const float*)d_in[7];
  const float* W_ho   = (const float*)d_in[8];
  const float* W_xc   = (const float*)d_in[9];
  const float* W_hc   = (const float*)d_in[10];
  const float* b_i    = (const float*)d_in[11];
  const float* b_f    = (const float*)d_in[12];
  const float* b_o    = (const float*)d_in[13];
  const float* b_c    = (const float*)d_in[14];

  // Workspace: bf16 weight repack, 1024*512*2 = 1 MiB
  unsigned short* Wt = (unsigned short*)d_ws;

  convert_weights<<<dim3((4 * HID * KDIM) / 256), 256, 0, stream>>>(
      W_xi, W_hi, W_xf, W_hf, W_xo, W_ho, W_xc, W_hc, Wt);

  float* out_h = (float*)d_out;
  float* out_c = out_h + (size_t)BATCH * HID;

  dim3 grid(BATCH / 32);   // 2048 workgroups, each owns a full 32-row stripe
  lstm_wmma_kernel<<<grid, 256, 0, stream>>>(
      x, h_prev, c_prev, Wt, b_i, b_f, b_o, b_c, out_h, out_c);
}
